// MistralDecoder_2877628088449
// MI455X (gfx1250) — matrix operand order repaired and checked by reading, not compiled, not run on hardware
//
#include <hip/hip_runtime.h>
#include <cstdint>
#include <cstddef>

// ---------------- problem constants ----------------
#define B_   2
#define S_   2048
#define E_   1024
#define HQ_  16
#define HKV_ 4
#define D_   64
#define WIN_ 512
#define HID_ 4096
#define M_   (B_ * S_)        // 4096 token rows

typedef unsigned short u16;   // bf16 storage
typedef __attribute__((ext_vector_type(16))) __bf16 v16bf;
typedef __attribute__((ext_vector_type(8)))  float  v8f;
typedef __attribute__((ext_vector_type(4)))  int    v4i;

union BF16Frag { v16bf v; u16 u[16]; };

// ---- gfx1250 async global->LDS copy (ASYNCcnt path), probe-confirmed present ----
#if defined(__AMDGCN__) && __has_builtin(__builtin_amdgcn_global_load_async_to_lds_b128)
#define HAVE_ASYNC_LDS 1
typedef v4i __attribute__((address_space(1)))* gptr_v4i;   // global src (param 0)
typedef v4i __attribute__((address_space(3)))* lptr_v4i;   // LDS dst   (param 1)
#else
#define HAVE_ASYNC_LDS 0
#endif

__device__ __forceinline__ void wait_asynccnt0() {
#if defined(__AMDGCN__)
#if __has_builtin(__builtin_amdgcn_s_wait_asynccnt)
  __builtin_amdgcn_s_wait_asynccnt(0);
#else
  asm volatile("s_wait_asynccnt 0x0" ::: "memory");
#endif
#endif
}

__device__ __forceinline__ u16 f2bf(float f) {
  union { __bf16 h; u16 u; } cv; cv.h = (__bf16)f; return cv.u;
}

__device__ __forceinline__ v8f wmma_bf16(v16bf a, v16bf b, v8f c) {
  // D = A(16x32 bf16) * B(32x16 bf16) + C(16x16 f32)
  return __builtin_amdgcn_wmma_f32_16x16x32_bf16(
      /*neg_a=*/false, a, /*neg_b=*/false, b,
      /*c_mod=*/(short)0, c, /*reuse_a=*/false, /*reuse_b=*/false);
}

// A fragment (16x32, row-major source, leading dim ld in elements).
// lane<16 holds row=lane, K=0..15; lane>=16 row=lane-16, K=16..31
__device__ __forceinline__ v16bf load_a_frag(const u16* p, int ld, int lane) {
  const int half = lane >> 4, r = lane & 15;
  const u16* q = p + (size_t)r * ld + half * 16;
  BF16Frag f;
#pragma unroll
  for (int t = 0; t < 16; ++t) { f.u[t] = q[t]; }
  return f.v;
}

// B fragment (32x16) from a [K][N] (n-contiguous) buffer, leading dim ld:
// lane<16 holds col n=lane, K=0..15; lane>=16 col=lane-16, K=16..31 (strided reads)
__device__ __forceinline__ v16bf load_b_frag_kstrided(const u16* p, int ld, int lane) {
  const int half = lane >> 4, n = lane & 15;
  const u16* q = p + (size_t)(half * 16) * ld + n;
  BF16Frag f;
#pragma unroll
  for (int t = 0; t < 16; ++t) f.u[t] = q[(size_t)t * ld];
  return f.v;
}

// ---------------- elementwise kernels ----------------
__global__ __launch_bounds__(256) void cast_f32_bf16_kernel(
    const float* __restrict__ in, u16* __restrict__ out, int n) {
  for (int i = blockIdx.x * 256 + threadIdx.x; i < n; i += gridDim.x * 256)
    out[i] = f2bf(in[i]);
}

__global__ __launch_bounds__(256) void rmsnorm_kernel(
    const float* __restrict__ x, const float* __restrict__ g,
    u16* __restrict__ out, int ncols) {
  const int row = blockIdx.x;
  const float* xr = x + (size_t)row * ncols;
  float ss = 0.0f;
  for (int c = threadIdx.x; c < ncols; c += 256) { float v = xr[c]; ss += v * v; }
#pragma unroll
  for (int o = 16; o > 0; o >>= 1) ss += __shfl_xor(ss, o, 32);
  __shared__ float red[8];
  __shared__ float stot;
  if ((threadIdx.x & 31) == 0) red[threadIdx.x >> 5] = ss;
  __syncthreads();
  if (threadIdx.x == 0) {
    float t = 0.0f;
#pragma unroll
    for (int i = 0; i < 8; ++i) t += red[i];
    stot = rsqrtf(t / (float)ncols + 1e-6f);
  }
  __syncthreads();
  const float scale = stot;
  u16* orow = out + (size_t)row * ncols;
  for (int c = threadIdx.x; c < ncols; c += 256)
    orow[c] = f2bf(xr[c] * scale * g[c]);
}

// RoPE on fp32 [B,S,H,D] rows, cast to bf16. One thread per (b,s,h,d') pair, d'<32.
__global__ __launch_bounds__(256) void rope_cast_kernel(
    const float* __restrict__ in, u16* __restrict__ out, int H, int total) {
  int idx = blockIdx.x * 256 + threadIdx.x;
  if (idx >= total) return;
  const int dp = idx & 31;
  const int rest = idx >> 5;
  const int h = rest % H;
  const int bs = rest / H;
  const int pos = bs % S_;
  const size_t base = ((size_t)bs * H + h) * D_;
  const float inv = __powf(10000.0f, -(float)dp * (1.0f / 32.0f));
  float sn, cs;
  __sincosf((float)pos * inv, &sn, &cs);
  const float x1 = in[base + dp];
  const float x2 = in[base + dp + 32];
  out[base + dp]      = f2bf(x1 * cs - x2 * sn);
  out[base + dp + 32] = f2bf(x2 * cs + x1 * sn);
}

// ---------------- tiled WMMA GEMM ----------------
// C[M,N] = A[M,K](bf16,row-major) * B[K,N](bf16,row-major); f32 accumulate.
// Block tile 64x256x64, 8 waves: wave = (wm in 0..1 -> 32 rows) x (wn in 0..3 -> 64 cols).
// Each wave: 2 A-frags x 4 B-frags x 2 K-slices = 16 wmma per LDS stage.
enum { EP_F32 = 0, EP_RES_F32 = 1, EP_BF16 = 2, EP_SILU_MUL_BF16 = 3 };

template <int EPI>
__global__ __launch_bounds__(256)
void gemm_bf16_kernel(const u16* __restrict__ A, const u16* __restrict__ Bw,
                      int M, int N, int K,
                      float* __restrict__ Cf, u16* __restrict__ Cb,
                      const float* __restrict__ aux) {
  constexpr int BM = 64, BN = 256, BK = 64;
  __shared__ u16 As[BM][BK];     // [m][k] natural order
  __shared__ u16 Bs[BK][BN];     // [k][n] natural order

  const int tid  = threadIdx.x;
  const int lane = tid & 31;
  const int wave = tid >> 5;
  const int wm = wave & 1;             // 2 wave-slots along M (32 rows each)
  const int wn = wave >> 1;            // 4 wave-slots along N (64 cols each)
  const int bm = blockIdx.x * BM;
  const int bn = blockIdx.y * BN;

  // A staging: 64x64 = 512 16B-chunks -> 2 per thread
  const int ar = tid >> 3;             // 0..31 (and +32)
  const int ac = (tid & 7) * 8;
  // B staging: 64x256 = 2048 16B-chunks -> 8 per thread
  const int br = tid >> 5;             // 0..7 (and +8..+56)
  const int bc = (tid & 31) * 8;

  v8f acc[2][4];
#pragma unroll
  for (int i = 0; i < 2; ++i)
#pragma unroll
    for (int j = 0; j < 4; ++j) acc[i][j] = v8f{};

  for (int k0 = 0; k0 < K; k0 += BK) {
#if HAVE_ASYNC_LDS
    // ASYNCcnt-tracked direct global->LDS b128 copies (no VGPR round trip)
#pragma unroll
    for (int i = 0; i < 2; ++i) {
      u16* gsrc = (u16*)&A[(size_t)(bm + ar + 32 * i) * K + k0 + ac];
      __builtin_amdgcn_global_load_async_to_lds_b128(
          (gptr_v4i)gsrc, (lptr_v4i)&As[ar + 32 * i][ac], 0, 0);
    }
#pragma unroll
    for (int i = 0; i < 8; ++i) {
      u16* gsrc = (u16*)&Bw[(size_t)(k0 + br + 8 * i) * N + bn + bc];
      __builtin_amdgcn_global_load_async_to_lds_b128(
          (gptr_v4i)gsrc, (lptr_v4i)&Bs[br + 8 * i][bc], 0, 0);
    }
    wait_asynccnt0();
#else
#pragma unroll
    for (int i = 0; i < 2; ++i)
      *(uint4*)&As[ar + 32 * i][ac] =
          *(const uint4*)&A[(size_t)(bm + ar + 32 * i) * K + k0 + ac];
#pragma unroll
    for (int i = 0; i < 8; ++i)
      *(uint4*)&Bs[br + 8 * i][bc] =
          *(const uint4*)&Bw[(size_t)(k0 + br + 8 * i) * N + bn + bc];
    if (k0 + BK < K) {   // hint next tiles (global_prefetch_b8)
      __builtin_prefetch(&A[(size_t)(bm + ar) * K + k0 + BK + ac], 0, 0);
      __builtin_prefetch(&Bw[(size_t)(k0 + BK + br) * N + bn + bc], 0, 0);
    }
#endif
    __syncthreads();

#pragma unroll
    for (int ks = 0; ks < BK; ks += 32) {
      const v16bf a0 = load_a_frag(&As[wm * 32 +  0][ks], BK, lane);
      const v16bf a1 = load_a_frag(&As[wm * 32 + 16][ks], BK, lane);
#pragma unroll
      for (int nf = 0; nf < 4; ++nf) {
        const v16bf bfr =
            load_b_frag_kstrided(&Bs[ks][wn * 64 + nf * 16], BN, lane);
        acc[0][nf] = wmma_bf16(a0, bfr, acc[0][nf]);
        acc[1][nf] = wmma_bf16(a1, bfr, acc[1][nf]);
      }
    }
    __syncthreads();
  }

  // epilogue: C layout — VGPR r holds row r (lanes 0-15) / row r+8 (lanes 16-31)
  const int half = lane >> 4;
  const int nl   = lane & 15;
#pragma unroll
  for (int mf = 0; mf < 2; ++mf) {
#pragma unroll
    for (int nf = 0; nf < 4; ++nf) {
      const int col = bn + wn * 64 + nf * 16 + nl;
#pragma unroll
      for (int r = 0; r < 8; ++r) {
        const int row = bm + wm * 32 + mf * 16 + r + 8 * half;
        const size_t idx = (size_t)row * N + col;
        const float v = acc[mf][nf][r];
        if (EPI == EP_F32) {
          Cf[idx] = v;
        } else if (EPI == EP_RES_F32) {
          Cf[idx] = v + aux[idx];
        } else if (EPI == EP_BF16) {
          Cb[idx] = f2bf(v);
        } else { // EP_SILU_MUL_BF16: out = bf16(silu(aux) * v)
          const float g = aux[idx];
          const float sg = g * (1.0f / (1.0f + __expf(-g)));
          Cb[idx] = f2bf(sg * v);
        }
      }
    }
  }
}

// ---------------- sliding-window GQA attention ----------------
// One wave per (b, q-head, 16-query tile). Q*K^T via WMMA, fp32 softmax,
// P staged in LDS (A-fragment friendly), P*V via WMMA. GQA: hk = h/4.
#define KMAX 544   // max key span: WIN + tile padding, multiple of 32

__global__ __launch_bounds__(32)
void attn_kernel(const u16* __restrict__ qb, const u16* __restrict__ kb,
                 const u16* __restrict__ vb, u16* __restrict__ ctxb) {
  __shared__ float sc[16][KMAX];   // raw scores
  __shared__ u16   pb[16][KMAX];   // exp(s - max) as bf16 (unnormalized)
  __shared__ float sinv[16];       // 1/rowsum

  const int lane = threadIdx.x;
  const int bid  = blockIdx.x;
  const int qt = bid % (S_ / 16);
  const int h  = (bid / (S_ / 16)) % HQ_;
  const int b  = bid / ((S_ / 16) * HQ_);
  const int i0 = qt * 16;
  const int hk = h / (HQ_ / HKV_);

  const u16* qbase = qb + (size_t)b * S_ * HQ_ * D_ + (size_t)h * D_;   // stride HQ_*D_ per i
  const u16* kbase = kb + (size_t)b * S_ * HKV_ * D_ + (size_t)hk * D_; // stride HKV_*D_ per j
  const u16* vbase = vb + (size_t)b * S_ * HKV_ * D_ + (size_t)hk * D_;

  // Q fragments for d-slices [0,32) and [32,64)
  const v16bf qf0 = load_a_frag(qbase + (size_t)i0 * (HQ_ * D_) + 0,  HQ_ * D_, lane);
  const v16bf qf1 = load_a_frag(qbase + (size_t)i0 * (HQ_ * D_) + 32, HQ_ * D_, lane);

  const int jstart = (i0 - (WIN_ - 1)) > 0 ? (i0 - (WIN_ - 1)) : 0;
  const int jt0 = jstart & ~31;
  const int jend = i0 + 15;
  const int nt = (jend - jt0 + 32) / 32;   // <= 17 key tiles of 32

  const int half = lane >> 4;
  const int nl   = lane & 15;

  // ---- scores: S = Q K^T / sqrt(D), masked ----
  for (int t = 0; t < nt; ++t) {
    const int jt = jt0 + t * 32;
#pragma unroll
    for (int hh = 0; hh < 2; ++hh) {
      const int col = jt + hh * 16 + nl;           // key index for this lane
      const int jc  = col < (S_ - 1) ? col : (S_ - 1);
      const u16* kp = kbase + (size_t)jc * (HKV_ * D_) + half * 16;
      BF16Frag b0, b1;
#pragma unroll
      for (int e = 0; e < 16; ++e) { b0.u[e] = kp[e]; b1.u[e] = kp[32 + e]; }
      v8f acc = {};
      acc = wmma_bf16(qf0, b0.v, acc);
      acc = wmma_bf16(qf1, b1.v, acc);
#pragma unroll
      for (int r = 0; r < 8; ++r) {
        const int m = r + 8 * half;
        const int i = i0 + m;
        const bool ok = (col <= i) && (col > i - WIN_);
        sc[m][col - jt0] = ok ? acc[r] * 0.125f : -1e30f;
      }
    }
  }
  __syncthreads();

  // ---- softmax (fp32): lanes 0..15 even cols, 16..31 odd cols of row lane&15 ----
  {
    const int row = lane & 15, part = lane >> 4;
    const int nk = nt * 32;
    float mx = -1e30f;
    for (int c = part; c < nk; c += 2) mx = fmaxf(mx, sc[row][c]);
    mx = fmaxf(mx, __shfl_xor(mx, 16, 32));
    float sum = 0.0f;
    for (int c = part; c < nk; c += 2) {
      const float e = __expf(sc[row][c] - mx);
      sum += e;
      pb[row][c] = f2bf(e);
    }
    sum += __shfl_xor(sum, 16, 32);
    if (lane < 16) sinv[row] = 1.0f / sum;
  }
  __syncthreads();

  // ---- ctx = P V (unnormalized), scaled by 1/rowsum at store ----
  v8f acc[4] = {v8f{}, v8f{}, v8f{}, v8f{}};
  for (int t = 0; t < nt; ++t) {
    const v16bf pf = load_a_frag(&pb[0][t * 32], KMAX, lane);
    const int kjb = jt0 + t * 32 + half * 16;
#pragma unroll
    for (int f = 0; f < 4; ++f) {
      const u16* vp = vbase + f * 16 + nl;     // col d = f*16 + nl, keys strided
      BF16Frag bfv;
#pragma unroll
      for (int e = 0; e < 16; ++e) {
        const int j  = kjb + e;
        const int jc = j < (S_ - 1) ? j : (S_ - 1);
        bfv.u[e] = vp[(size_t)jc * (HKV_ * D_)];
      }
      acc[f] = wmma_bf16(pf, bfv.v, acc[f]);
    }
  }

  u16* cbase = ctxb + (size_t)b * S_ * HQ_ * D_ + (size_t)h * D_;
#pragma unroll
  for (int f = 0; f < 4; ++f)
#pragma unroll
    for (int r = 0; r < 8; ++r) {
      const int m = r + 8 * half;
      cbase[(size_t)(i0 + m) * (HQ_ * D_) + f * 16 + nl] = f2bf(acc[f][r] * sinv[m]);
    }
}

// ---------------- host orchestration ----------------
extern "C" void kernel_launch(void* const* d_in, const int* in_sizes, int n_in,
                              void* d_out, int out_size, void* d_ws, size_t ws_size,
                              hipStream_t stream) {
  (void)in_sizes; (void)n_in; (void)out_size; (void)ws_size;
  const float* x   = (const float*)d_in[0];
  const float* wq  = (const float*)d_in[1];
  const float* wk  = (const float*)d_in[2];
  const float* wv  = (const float*)d_in[3];
  const float* wo  = (const float*)d_in[4];
  const float* wg  = (const float*)d_in[5];
  const float* wu  = (const float*)d_in[6];
  const float* wd  = (const float*)d_in[7];
  const float* g1v = (const float*)d_in[8];
  const float* g2v = (const float*)d_in[9];
  float* out = (float*)d_out;

  char* ws = (char*)d_ws;
  size_t off = 0;
  auto alloc = [&](size_t bytes) -> void* {
    void* p = ws + off;
    off += (bytes + 255) & ~(size_t)255;
    return p;
  };

  u16*   wq_bf = (u16*)alloc((size_t)E_ * HQ_ * D_ * 2);
  u16*   wk_bf = (u16*)alloc((size_t)E_ * HKV_ * D_ * 2);
  u16*   wv_bf = (u16*)alloc((size_t)E_ * HKV_ * D_ * 2);
  u16*   wo_bf = (u16*)alloc((size_t)HQ_ * D_ * E_ * 2);
  u16*   wg_bf = (u16*)alloc((size_t)E_ * HID_ * 2);
  u16*   wu_bf = (u16*)alloc((size_t)E_ * HID_ * 2);
  u16*   wd_bf = (u16*)alloc((size_t)HID_ * E_ * 2);
  u16*   hb    = (u16*)alloc((size_t)M_ * E_ * 2);
  float* qf    = (float*)alloc((size_t)M_ * HQ_ * D_ * 4);
  float* kf    = (float*)alloc((size_t)M_ * HKV_ * D_ * 4);
  u16*   qbb   = (u16*)alloc((size_t)M_ * HQ_ * D_ * 2);
  u16*   kbb   = (u16*)alloc((size_t)M_ * HKV_ * D_ * 2);
  u16*   vbb   = (u16*)alloc((size_t)M_ * HKV_ * D_ * 2);
  u16*   ctxb  = (u16*)alloc((size_t)M_ * HQ_ * D_ * 2);
  float* out1  = (float*)alloc((size_t)M_ * E_ * 4);
  u16*   h2b   = (u16*)alloc((size_t)M_ * E_ * 2);
  float* gbuf  = (float*)alloc((size_t)M_ * HID_ * 4);
  u16*   actb  = (u16*)alloc((size_t)M_ * HID_ * 2);

  // 1. weights -> bf16
  cast_f32_bf16_kernel<<<2048, 256, 0, stream>>>(wq, wq_bf, E_ * HQ_ * D_);
  cast_f32_bf16_kernel<<<1024, 256, 0, stream>>>(wk, wk_bf, E_ * HKV_ * D_);
  cast_f32_bf16_kernel<<<1024, 256, 0, stream>>>(wv, wv_bf, E_ * HKV_ * D_);
  cast_f32_bf16_kernel<<<2048, 256, 0, stream>>>(wo, wo_bf, HQ_ * D_ * E_);
  cast_f32_bf16_kernel<<<4096, 256, 0, stream>>>(wg, wg_bf, E_ * HID_);
  cast_f32_bf16_kernel<<<4096, 256, 0, stream>>>(wu, wu_bf, E_ * HID_);
  cast_f32_bf16_kernel<<<4096, 256, 0, stream>>>(wd, wd_bf, HID_ * E_);

  // 2. h = rmsnorm(x, g1) -> bf16
  rmsnorm_kernel<<<M_, 256, 0, stream>>>(x, g1v, hb, E_);

  // 3. QKV projections (WMMA bf16, f32 accumulate)
  gemm_bf16_kernel<EP_F32><<<dim3(M_ / 64, (HQ_ * D_) / 256), 256, 0, stream>>>(
      hb, wq_bf, M_, HQ_ * D_, E_, qf, nullptr, nullptr);
  gemm_bf16_kernel<EP_F32><<<dim3(M_ / 64, (HKV_ * D_) / 256), 256, 0, stream>>>(
      hb, wk_bf, M_, HKV_ * D_, E_, kf, nullptr, nullptr);
  gemm_bf16_kernel<EP_BF16><<<dim3(M_ / 64, (HKV_ * D_) / 256), 256, 0, stream>>>(
      hb, wv_bf, M_, HKV_ * D_, E_, nullptr, vbb, nullptr);

  // 4. RoPE (fp32) + cast to bf16
  rope_cast_kernel<<<(M_ * HQ_ * 32 + 255) / 256, 256, 0, stream>>>(
      qf, qbb, HQ_, M_ * HQ_ * 32);
  rope_cast_kernel<<<(M_ * HKV_ * 32 + 255) / 256, 256, 0, stream>>>(
      kf, kbb, HKV_, M_ * HKV_ * 32);

  // 5. sliding-window GQA attention
  attn_kernel<<<B_ * HQ_ * (S_ / 16), 32, 0, stream>>>(qbb, kbb, vbb, ctxb);

  // 6. out1 = ctx @ wo + x
  gemm_bf16_kernel<EP_RES_F32><<<dim3(M_ / 64, E_ / 256), 256, 0, stream>>>(
      ctxb, wo_bf, M_, E_, HQ_ * D_, out1, nullptr, x);

  // 7. h2 = rmsnorm(out1, g2)
  rmsnorm_kernel<<<M_, 256, 0, stream>>>(out1, g2v, h2b, E_);

  // 8. FFN: gate = h2@wg ; act = silu(gate) * (h2@wu) ; out = act@wd + out1
  gemm_bf16_kernel<EP_F32><<<dim3(M_ / 64, HID_ / 256), 256, 0, stream>>>(
      h2b, wg_bf, M_, HID_, E_, gbuf, nullptr, nullptr);
  gemm_bf16_kernel<EP_SILU_MUL_BF16><<<dim3(M_ / 64, HID_ / 256), 256, 0, stream>>>(
      h2b, wu_bf, M_, HID_, E_, nullptr, actb, gbuf);
  gemm_bf16_kernel<EP_RES_F32><<<dim3(M_ / 64, E_ / 256), 256, 0, stream>>>(
      actb, wd_bf, M_, E_, HID_, out, nullptr, out1);
}
